// SimpleMHA_17437567222121
// MI455X (gfx1250) — matrix operand order repaired and checked by reading, not compiled, not run on hardware
//
#include <hip/hip_runtime.h>
#include <hip/hip_bf16.h>

// ---------------------------------------------------------------------------
// SimpleMHA on gfx1250: fused QKV proj -> flash attention -> output proj.
// GEMMs: v_wmma_f32_16x16x32_f16.  Flash attention stages K/V^T tiles into
// LDS with the Tensor Data Mover (tensor_load_to_lds, TENSORcnt), double
// buffered, so the 8 waves of a block share one copy of each tile.
// ---------------------------------------------------------------------------

typedef __attribute__((ext_vector_type(16))) _Float16 v16h;
typedef __attribute__((ext_vector_type(8)))  _Float16 v8h;
typedef __attribute__((ext_vector_type(8)))  float    v8f;
typedef __attribute__((ext_vector_type(4)))  unsigned int u32x4;
typedef __attribute__((ext_vector_type(4)))  int      i32x4;
typedef __attribute__((ext_vector_type(8)))  int      i32x8;

#define EMBED   512
#define SEQ     4096
#define NHEAD   8
#define HDIM    64
#define BATCH   2
#define MROWS   (BATCH * SEQ)      // 8192
#define SCALE_F 0.125f             // 1/sqrt(64)
#define BK      64                 // keys staged per TDM block

#if defined(__has_include)
#if __has_include(<hip/amd_detail/amd_gfx1250_TDM.h>)
#define TDM_SIX_ARGS 1
#endif
#endif

__device__ __forceinline__ v8f zero8() {
    v8f z;
#pragma unroll
    for (int i = 0; i < 8; i++) z[i] = 0.0f;
    return z;
}

__device__ __forceinline__ v8f wmma_f16(v16h a, v16h b, v8f c) {
    return __builtin_amdgcn_wmma_f32_16x16x32_f16(
        false, a, false, b, (short)0, c, false, false);
}

// A-operand (16x32, M x K), row-major f16 source, row = row0 + lane%16.
// lane<16: K = c0+0..15 ; lane>=16: K = c0+16..31.
__device__ __forceinline__ v16h load_a_h(const _Float16* base, int ld,
                                         int row0, int c0, int lane) {
    const int hi = (lane >> 4) & 1, lo = lane & 15;
    const _Float16* p = base + (size_t)(row0 + lo) * ld + c0 + hi * 16;
    union { v16h v; v8h h[2]; } u;
    u.h[0] = *reinterpret_cast<const v8h*>(p);
    u.h[1] = *reinterpret_cast<const v8h*>(p + 8);
    return u.v;
}

// B-operand (32x16, K x N), column n = row (n0 + lane%16) of row-major src.
__device__ __forceinline__ v16h load_b_h(const _Float16* base, int ld,
                                         int n0, int c0, int lane) {
    const int hi = (lane >> 4) & 1, lo = lane & 15;
    const _Float16* p = base + (size_t)(n0 + lo) * ld + c0 + hi * 16;
    union { v16h v; v8h h[2]; } u;
    u.h[0] = *reinterpret_cast<const v8h*>(p);
    u.h[1] = *reinterpret_cast<const v8h*>(p + 8);
    return u.v;
}

__device__ __forceinline__ v16h load_a_f(const float* base, int ld,
                                         int row0, int c0, int lane) {
    const int hi = (lane >> 4) & 1, lo = lane & 15;
    const float* p = base + (size_t)(row0 + lo) * ld + c0 + hi * 16;
    v16h v;
#pragma unroll
    for (int i = 0; i < 8; i++) {
        v[i]     = (_Float16)p[i];
        v[8 + i] = (_Float16)p[8 + i];
    }
    return v;
}

__device__ __forceinline__ v16h load_b_f(const float* base, int ld,
                                         int n0, int c0, int lane) {
    const int hi = (lane >> 4) & 1, lo = lane & 15;
    const float* p = base + (size_t)(n0 + lo) * ld + c0 + hi * 16;
    v16h v;
#pragma unroll
    for (int i = 0; i < 16; i++) v[i] = (_Float16)p[i];
    return v;
}

// ---------------------------------------------------------------------------
// Tensor Data Mover: async 2-D tile load Global -> LDS (f16 elements).
// Descriptor per CDNA5 ISA ch.8: tensor declared == tile region (no OOB),
// dim0 = contiguous row length, stride0 = row pitch (elements).
// ---------------------------------------------------------------------------
__device__ __forceinline__ void tdm_load_2d(unsigned int lds_addr,
                                            const void* gptr,
                                            unsigned int dim0,
                                            unsigned int dim1,
                                            unsigned long long stride0) {
    unsigned long long ga = (unsigned long long)(uintptr_t)gptr;
    u32x4 g0;
    g0[0] = 1u;                                        // count=1, user mode
    g0[1] = lds_addr;                                  // LDS byte address
    g0[2] = (unsigned int)ga;                          // global_addr[31:0]
    g0[3] = (unsigned int)((ga >> 32) & 0x1FFFFFFu) | (2u << 30); // type=2
    i32x8 g1;
    g1[0] = (int)(1u << 16);                           // data_size=1 (2 bytes)
    g1[1] = (int)((dim0 & 0xFFFFu) << 16);             // tensor_dim0 lo16
    g1[2] = (int)(((dim0 >> 16) & 0xFFFFu) |
                  ((dim1 & 0xFFFFu) << 16));           // dim0 hi / dim1 lo
    g1[3] = (int)(((dim1 >> 16) & 0xFFFFu) |
                  ((dim0 & 0xFFFFu) << 16));           // dim1 hi / tile_dim0
    g1[4] = (int)(dim1 & 0xFFFFu);                     // tile_dim1; tile_dim2=0
    g1[5] = (int)(unsigned int)(stride0 & 0xFFFFFFFFull);
    g1[6] = (int)(unsigned int)((stride0 >> 32) & 0xFFFFull);
    g1[7] = 0;
    i32x4 z4; z4[0] = 0; z4[1] = 0; z4[2] = 0; z4[3] = 0;
#ifdef TDM_SIX_ARGS
    i32x8 z8;
#pragma unroll
    for (int i = 0; i < 8; i++) z8[i] = 0;
    __builtin_amdgcn_tensor_load_to_lds(g0, g1, z4, z4, z8, 0);
#else
    __builtin_amdgcn_tensor_load_to_lds(g0, g1, z4, z4, 0);
#endif
}

__device__ __forceinline__ unsigned int lds_addr_of(const void* p) {
    // Flat LDS address: high 32 bits = aperture, low 32 bits = LDS byte offset.
    return (unsigned int)(uintptr_t)p;
}

// ---------------------------------------------------------------------------
// Kernel 1: fused QKV projection (scale folded into Q; V stored transposed).
// ---------------------------------------------------------------------------
__global__ __launch_bounds__(256) void qkv_proj_kernel(
    const float* __restrict__ x,
    const float* __restrict__ Wq, const float* __restrict__ bq,
    const float* __restrict__ Wk, const float* __restrict__ bk,
    const float* __restrict__ Wv, const float* __restrict__ bv,
    _Float16* __restrict__ Qh, _Float16* __restrict__ Kh,
    _Float16* __restrict__ Vt)
{
    const int lane  = threadIdx.x & 31;
    const int wave  = threadIdx.x >> 5;
    const int m0    = blockIdx.x * 16;
    const int which = blockIdx.y;               // 0=Q, 1=K, 2=V
    const int n0    = wave * 64;

    const float* W    = (which == 0) ? Wq : ((which == 1) ? Wk : Wv);
    const float* bias = (which == 0) ? bq : ((which == 1) ? bk : bv);

    v8f acc[4];
#pragma unroll
    for (int t = 0; t < 4; t++) acc[t] = zero8();

    for (int c0 = 0; c0 < EMBED; c0 += 32) {
        v16h a = load_a_f(x, EMBED, m0, c0, lane);
#pragma unroll
        for (int t = 0; t < 4; t++) {
            v16h b = load_b_f(W, EMBED, n0 + t * 16, c0, lane);
            acc[t] = wmma_f16(a, b, acc[t]);
        }
    }

    const int hi = lane >> 4, lo = lane & 15;
    const float scale = (which == 0) ? SCALE_F : 1.0f;
#pragma unroll
    for (int t = 0; t < 4; t++) {
        const int n  = n0 + t * 16 + lo;
        const float bn = bias[n];
        const int h = n >> 6, d = n & 63;
#pragma unroll
        for (int r = 0; r < 8; r++) {
            const int m   = m0 + hi * 8 + r;
            const int b   = m >> 12;
            const int tok = m & 4095;
            const float val = (acc[t][r] + bn) * scale;
            if (which == 2) {
                Vt[((size_t)(b * NHEAD + h) * HDIM + d) * SEQ + tok] =
                    (_Float16)val;
            } else {
                const size_t idx =
                    ((size_t)(b * NHEAD + h) * SEQ + tok) * HDIM + d;
                if (which == 0) Qh[idx] = (_Float16)val;
                else            Kh[idx] = (_Float16)val;
            }
        }
    }
}

// ---------------------------------------------------------------------------
// Kernel 2: flash attention, TDM double-buffered K/V^T staging in LDS.
//   S^T = K * Q^T (softmax stats per-lane), O^T = V^T * P^T.
//   block = 8 waves = 128 queries; grid = (SEQ/128, B*H).
// ---------------------------------------------------------------------------
__global__ __launch_bounds__(256) void flash_attn_kernel(
    const _Float16* __restrict__ Qh, const _Float16* __restrict__ Kh,
    const _Float16* __restrict__ Vt, _Float16* __restrict__ Oh)
{
    __shared__ __align__(16) _Float16 sK[2][BK][HDIM];   // 2 x 8 KB
    __shared__ __align__(16) _Float16 sV[2][HDIM][BK];   // 2 x 8 KB

    const int lane = threadIdx.x & 31;
    const int wave = threadIdx.x >> 5;
    const int bh   = blockIdx.y;
    const int q0   = blockIdx.x * 128 + wave * 16;

    const _Float16* Qb = Qh + (size_t)bh * SEQ * HDIM;
    const _Float16* Kb = Kh + (size_t)bh * SEQ * HDIM;
    const _Float16* Vb = Vt + (size_t)bh * HDIM * SEQ;

    // Q as B-operand, resident for the whole kernel.
    const v16h qf0 = load_b_h(Qb, HDIM, q0, 0,  lane);
    const v16h qf1 = load_b_h(Qb, HDIM, q0, 32, lane);

    v8f ot[4];
#pragma unroll
    for (int t = 0; t < 4; t++) ot[t] = zero8();
    float m_i = -3.0e38f, l_i = 0.0f;
    const int hi = lane >> 4;

    // Prologue: TDM-stage block 0 into buffer 0.
    if (wave == 0) {
        tdm_load_2d(lds_addr_of(&sK[0][0][0]), Kb, HDIM, BK, HDIM);
        tdm_load_2d(lds_addr_of(&sV[0][0][0]), Vb, BK, HDIM, SEQ);
        __builtin_amdgcn_s_wait_tensorcnt(0);
    }
    __syncthreads();

    const int NBLK = SEQ / BK;
    for (int j = 0; j < NBLK; ++j) {
        const int buf = j & 1;
        // Prefetch block j+1 into the other buffer (overlaps with compute).
        if (wave == 0 && (j + 1) < NBLK) {
            const int kn = (j + 1) * BK;
            tdm_load_2d(lds_addr_of(&sK[buf ^ 1][0][0]),
                        Kb + (size_t)kn * HDIM, HDIM, BK, HDIM);
            tdm_load_2d(lds_addr_of(&sV[buf ^ 1][0][0]),
                        Vb + kn, BK, HDIM, SEQ);
        }

        const _Float16* Ks = &sK[buf][0][0];
        const _Float16* Vs = &sV[buf][0][0];

#pragma unroll
        for (int kk = 0; kk < BK; kk += 32) {
            v8f T0 = zero8(), T1 = zero8();
            T0 = wmma_f16(load_a_h(Ks, HDIM, kk,      0,  lane), qf0, T0);
            T0 = wmma_f16(load_a_h(Ks, HDIM, kk,      32, lane), qf1, T0);
            T1 = wmma_f16(load_a_h(Ks, HDIM, kk + 16, 0,  lane), qf0, T1);
            T1 = wmma_f16(load_a_h(Ks, HDIM, kk + 16, 32, lane), qf1, T1);

            float mx = T0[0];
#pragma unroll
            for (int r = 0; r < 8; r++) {
                mx = fmaxf(mx, T0[r]);
                mx = fmaxf(mx, T1[r]);
            }
            mx = fmaxf(mx, __shfl_xor(mx, 16, 32));
            const float m_new = fmaxf(m_i, mx);
            const float alpha = __expf(m_i - m_new);

            float p0[8], p1[8], s = 0.0f;
#pragma unroll
            for (int r = 0; r < 8; r++) {
                p0[r] = __expf(T0[r] - m_new);
                p1[r] = __expf(T1[r] - m_new);
                s += p0[r] + p1[r];
            }
            s += __shfl_xor(s, 16, 32);
            l_i = l_i * alpha + s;
            m_i = m_new;
#pragma unroll
            for (int t = 0; t < 4; t++) ot[t] = ot[t] * alpha;

            // Build P^T B-operand via one half-wave exchange.
            v16h pb;
#pragma unroll
            for (int r = 0; r < 8; r++) {
                const float x0 = __shfl_xor(p0[r], 16, 32);
                const float x1 = __shfl_xor(p1[r], 16, 32);
                pb[r]     = (_Float16)(hi ? x1 : p0[r]);
                pb[8 + r] = (_Float16)(hi ? p1[r] : x0);
            }

            ot[0] = wmma_f16(load_a_h(Vs, BK, 0,  kk, lane), pb, ot[0]);
            ot[1] = wmma_f16(load_a_h(Vs, BK, 16, kk, lane), pb, ot[1]);
            ot[2] = wmma_f16(load_a_h(Vs, BK, 32, kk, lane), pb, ot[2]);
            ot[3] = wmma_f16(load_a_h(Vs, BK, 48, kk, lane), pb, ot[3]);
        }

        // Next buffer must have landed before anyone proceeds.
        if (wave == 0) __builtin_amdgcn_s_wait_tensorcnt(0);
        __syncthreads();
    }

    // Normalize and write f16 [B*N, 512] rows (feature = h*64 + d).
    const float rl = 1.0f / l_i;
    const int lo = lane & 15;
    const int b = bh >> 3, h = bh & 7;
    const int n = q0 + lo;
    _Float16* orow = Oh + ((size_t)(b * SEQ + n)) * EMBED + h * HDIM;
#pragma unroll
    for (int t = 0; t < 4; t++) {
        v8h ov;
#pragma unroll
        for (int r = 0; r < 8; r++) ov[r] = (_Float16)(ot[t][r] * rl);
        *reinterpret_cast<v8h*>(orow + t * 16 + hi * 8) = ov;
    }
}

// ---------------------------------------------------------------------------
// Kernel 3: output projection  out = attn @ Wo^T + bo  (f32 result).
// ---------------------------------------------------------------------------
__global__ __launch_bounds__(256) void out_proj_kernel(
    const _Float16* __restrict__ Oh, const float* __restrict__ Wo,
    const float* __restrict__ bo, float* __restrict__ out)
{
    const int lane = threadIdx.x & 31;
    const int wave = threadIdx.x >> 5;
    const int m0   = blockIdx.x * 16;
    const int n0   = wave * 64;

    v8f acc[4];
#pragma unroll
    for (int t = 0; t < 4; t++) acc[t] = zero8();

    for (int c0 = 0; c0 < EMBED; c0 += 32) {
        v16h a = load_a_h(Oh, EMBED, m0, c0, lane);
#pragma unroll
        for (int t = 0; t < 4; t++) {
            v16h b = load_b_f(Wo, EMBED, n0 + t * 16, c0, lane);
            acc[t] = wmma_f16(a, b, acc[t]);
        }
    }

    const int hi = lane >> 4, lo = lane & 15;
#pragma unroll
    for (int t = 0; t < 4; t++) {
        const int n = n0 + t * 16 + lo;
        const float bn = bo[n];
#pragma unroll
        for (int r = 0; r < 8; r++) {
            const int m = m0 + hi * 8 + r;
            out[(size_t)m * EMBED + n] = acc[t][r] + bn;
        }
    }
}

// ---------------------------------------------------------------------------
extern "C" void kernel_launch(void* const* d_in, const int* in_sizes, int n_in,
                              void* d_out, int out_size, void* d_ws,
                              size_t ws_size, hipStream_t stream) {
    const float* x  = (const float*)d_in[0];
    const float* Wq = (const float*)d_in[1];
    const float* bq = (const float*)d_in[2];
    const float* Wk = (const float*)d_in[3];
    const float* bk = (const float*)d_in[4];
    const float* Wv = (const float*)d_in[5];
    const float* bv = (const float*)d_in[6];
    const float* Wo = (const float*)d_in[7];
    const float* bo = (const float*)d_in[8];
    float* out = (float*)d_out;

    const size_t elems = (size_t)MROWS * EMBED;
    _Float16* Qh = (_Float16*)d_ws;
    _Float16* Kh = Qh + elems;
    _Float16* Vt = Kh + elems;
    _Float16* Oh = Vt + elems;

    dim3 blk(256);
    qkv_proj_kernel<<<dim3(MROWS / 16, 3), blk, 0, stream>>>(
        x, Wq, bq, Wk, bk, Wv, bv, Qh, Kh, Vt);
    flash_attn_kernel<<<dim3(SEQ / 128, BATCH * NHEAD), blk, 0, stream>>>(
        Qh, Kh, Vt, Oh);
    out_proj_kernel<<<dim3(MROWS / 16, 1), blk, 0, stream>>>(
        Oh, Wo, bo, out);
}
